// MyMultiHeadedAttention_71176198029522
// MI455X (gfx1250) — matrix operand order repaired and checked by reading, not compiled, not run on hardware
//
#include <hip/hip_runtime.h>
#include <hip/hip_bf16.h>

// ---------------------------------------------------------------------------
// Multi-head attention for MI455X (gfx1250, wave32, WMMA 16x16x32 f16).
//   B=4, T1=T2=2048, N_FEAT=1024, H=16, D_K=64
// ---------------------------------------------------------------------------

typedef _Float16 v4h  __attribute__((ext_vector_type(4)));
typedef _Float16 v8h  __attribute__((ext_vector_type(8)));
typedef _Float16 v16h __attribute__((ext_vector_type(16)));
typedef float    v8f  __attribute__((ext_vector_type(8)));
typedef int      v4i  __attribute__((ext_vector_type(4)));

#define B_SZ  4
#define HEADS 16
#define T1L   2048
#define T2L   2048
#define DKH   64
#define NFEAT 1024

#if defined(__gfx1250__) && __has_builtin(__builtin_amdgcn_global_load_async_to_lds_b128)
#define HAVE_ASYNC_LDS 1
typedef __attribute__((address_space(1))) v4i* gptr_v4i;
typedef __attribute__((address_space(3))) v4i* lptr_v4i;
#endif

static __device__ __forceinline__ v16h cat16(v8h lo, v8h hi) {
  return __builtin_shufflevector(lo, hi, 0,1,2,3,4,5,6,7,8,9,10,11,12,13,14,15);
}
static __device__ __forceinline__ v8f wmma_f16(v16h a, v16h b, v8f c) {
  // (neg_a, A, neg_b, B, c_mod, C, reuse_a, reuse_b) -> v_wmma_f32_16x16x32_f16
  return __builtin_amdgcn_wmma_f32_16x16x32_f16(false, a, false, b, (short)0, c, false, false);
}
static __device__ __forceinline__ void wait_async_zero() {
#if defined(__gfx1250__)
#if __has_builtin(__builtin_amdgcn_s_wait_asynccnt)
  __builtin_amdgcn_s_wait_asynccnt(0);
#else
  asm volatile("s_wait_asynccnt 0" ::: "memory");
#endif
#endif
}
static __device__ __forceinline__ void stage_k_tile(const _Float16* kg, _Float16* lds) {
#ifdef HAVE_ASYNC_LDS
  // gfx1250 async DMA: global -> LDS without a VGPR round trip (ASYNCcnt).
  __builtin_amdgcn_global_load_async_to_lds_b128((gptr_v4i)(v4i*)kg,
                                                 (lptr_v4i)(v4i*)lds, 0, 0);
#else
  *(v8h*)lds = *(const v8h*)kg;
#endif
}

// ---------------------------------------------------------------------------
// Kernel 1a: f32 [B,T,1024] -> f16 head-major [B,H,T,64]
// ---------------------------------------------------------------------------
__global__ __launch_bounds__(256) void cvt_head_kernel(const float* __restrict__ src,
                                                       _Float16* __restrict__ dst) {
  const int idx4 = (blockIdx.x * 256 + threadIdx.x) * 4;
  const int b   = idx4 >> 21;            // / (2048*1024)
  const int rem = idx4 & ((1 << 21) - 1);
  const int t   = rem >> 10;             // / 1024
  const int f   = rem & 1023;
  const int h   = f >> 6;
  const int d   = f & 63;
  const float4 v = *(const float4*)(src + idx4);
  v4h o;
  o[0] = (_Float16)v.x; o[1] = (_Float16)v.y;
  o[2] = (_Float16)v.z; o[3] = (_Float16)v.w;
  const size_t pos = ((((size_t)b * HEADS + h) * T2L + t) << 6) + d;
  *(v4h*)(dst + pos) = o;
}

// ---------------------------------------------------------------------------
// Kernel 1b: plain f32 -> f16 copy (layout preserved). Used for query and Wq
// so the GEMM hot loop reads f16 directly (no per-iteration v_cvt, half bytes).
// ---------------------------------------------------------------------------
__global__ __launch_bounds__(256) void cvt_plain_kernel(const float* __restrict__ src,
                                                        _Float16* __restrict__ dst) {
  const int idx4 = (blockIdx.x * 256 + threadIdx.x) * 4;
  const float4 v = *(const float4*)(src + idx4);
  v4h o;
  o[0] = (_Float16)v.x; o[1] = (_Float16)v.y;
  o[2] = (_Float16)v.z; o[3] = (_Float16)v.w;
  *(v4h*)(dst + idx4) = o;
}

// ---------------------------------------------------------------------------
// Kernel 2: Q projection GEMM  y = x @ Wq^T + bq, all-f16 operands, f32 acc.
// Block = 128 threads (4 waves); block tile 64(M) x 64(N); K-step 32.
// Both A and B fragments are contiguous 2x16B f16 loads (K contiguous).
// ---------------------------------------------------------------------------
__global__ __launch_bounds__(128) void qproj_kernel(const _Float16* __restrict__ X,
                                                    const _Float16* __restrict__ W,
                                                    const float* __restrict__ bq,
                                                    _Float16* __restrict__ qh) {
  const int tid  = threadIdx.x;
  const int wave = tid >> 5;
  const int lane = tid & 31;
  const int hsel = lane >> 4;     // 0: lanes 0-15, 1: lanes 16-31
  const int r16  = lane & 15;

  const int mBase = blockIdx.x * 64 + wave * 16;
  const int gBase = blockIdx.y * 64;

  const _Float16* arow = X + (size_t)(mBase + r16) * NFEAT;
  const int k0 = hsel * 8;

  v8f acc[4] = {};
  for (int kb = 0; kb < NFEAT; kb += 32) {
    // A fragment (16x32): lane holds row r16, K = k0..k0+7 and 16+k0..+7
    const v16h a = cat16(*(const v8h*)(arow + kb + k0),
                         *(const v8h*)(arow + kb + 16 + k0));
#pragma unroll
    for (int j = 0; j < 4; ++j) {
      // B fragment (32x16): lane = output feature column, K contiguous in W row
      const _Float16* wr = W + (size_t)(gBase + j * 16 + r16) * NFEAT + kb + hsel * 8;
      const v16h bfr = cat16(*(const v8h*)wr, *(const v8h*)(wr + 16));
      acc[j] = wmma_f16(a, bfr, acc[j]);
    }
  }

  const int h = blockIdx.y;   // 64-wide N tile == exactly one head
#pragma unroll
  for (int j = 0; j < 4; ++j) {
    const float bias = bq[gBase + j * 16 + r16];
#pragma unroll
    for (int r = 0; r < 8; ++r) {
      const int m = mBase + r + 8 * hsel;       // C layout: VGPR r -> row r (+8 hi half)
      const int b = m >> 11;
      const int t = m & 2047;
      const int d = j * 16 + r16;
      qh[(((size_t)b * HEADS + h) * T1L + t) * DKH + d] = (_Float16)(acc[j][r] + bias);
    }
  }
}

// ---------------------------------------------------------------------------
// Kernel 3: flash attention. Block = 256 threads (8 waves) per (b,h,128 q-rows).
// Each wave: 16 q-rows x 64 d. Double-buffered 32-key tiles: the async K-DMA
// and the global V load for tile t+1 are issued BEFORE computing tile t, so
// the ASYNC engine overlaps with WMMA issue; one barrier per iteration.
// ---------------------------------------------------------------------------
__global__ __launch_bounds__(256) void attn_kernel(const _Float16* __restrict__ qh,
                                                   const _Float16* __restrict__ kh,
                                                   const _Float16* __restrict__ vh,
                                                   const int* __restrict__ mask,
                                                   float* __restrict__ out) {
  __shared__ _Float16 Kt[2][32][72];   // K tiles, row-major, padded
  __shared__ _Float16 Vt[2][64][40];   // V tiles transposed: Vt[buf][d][k]
  __shared__ _Float16 Pb[8][16][40];   // per-wave P buffer (16 q x 32 k)

  const int tid  = threadIdx.x;
  const int wave = tid >> 5;
  const int lane = tid & 31;
  const int hsel = lane >> 4;
  const int r16  = lane & 15;

  const int bh    = blockIdx.y;
  const int b     = bh >> 4;
  const int h     = bh & 15;
  const int qBase = blockIdx.x * 128;

  const _Float16* qptr  = qh + ((size_t)bh * T1L + qBase) * DKH;
  const _Float16* kptr  = kh + (size_t)bh * T2L * DKH;
  const _Float16* vptr  = vh + (size_t)bh * T2L * DKH;
  const int*      maskb = mask + b * T2L;

  // Q A-fragments for this wave's 16 rows (d 0..31 and 32..63), kept in VGPRs
  const int k0 = hsel * 16;
  const _Float16* qrow = qptr + (size_t)(wave * 16 + r16) * DKH;
  const v16h qa0 = cat16(*(const v8h*)(qrow + k0),      *(const v8h*)(qrow + 8 + k0));
  const v16h qa1 = cat16(*(const v8h*)(qrow + 32 + k0), *(const v8h*)(qrow + 40 + k0));

  float mrow[8], lrow[8];
  v8f O[4] = {};
#pragma unroll
  for (int r = 0; r < 8; ++r) { mrow[r] = -1e30f; lrow[r] = 0.0f; }

  const int elem  = tid * 8;          // cooperative loader mapping: 2048 halves / 256 thr
  const int ldRow = elem >> 6;        // 0..31 (key within tile)
  const int ldCol = elem & 63;        // d, multiple of 8

  constexpr int NT = T2L / 32;        // 64 key tiles

  // ---- prologue: fill buffer 0 with tile 0 ----
  {
    stage_k_tile(kptr + (size_t)ldRow * DKH + ldCol, &Kt[0][ldRow][ldCol]);
    const v8h vv = *(const v8h*)(vptr + (size_t)ldRow * DKH + ldCol);
#pragma unroll
    for (int i = 0; i < 8; ++i) Vt[0][ldCol + i][ldRow] = vv[i];
    wait_async_zero();
  }
  __syncthreads();

  for (int t = 0; t < NT; ++t) {
    const int cur = t & 1;
    const int nxt = cur ^ 1;
    const bool have_next = (t + 1 < NT);

    // ---- issue tile t+1 loads first: async K DMA + V global load into VGPRs ----
    v8h vv = {};
    if (have_next) {
      stage_k_tile(kptr + (size_t)((t + 1) * 32 + ldRow) * DKH + ldCol,
                   &Kt[nxt][ldRow][ldCol]);
      vv = *(const v8h*)(vptr + (size_t)((t + 1) * 32 + ldRow) * DKH + ldCol);
      if (t + 2 < NT) {
        __builtin_prefetch(kptr + (size_t)((t + 2) * 32 + ldRow) * DKH + ldCol, 0, 1);
        __builtin_prefetch(vptr + (size_t)((t + 2) * 32 + ldRow) * DKH + ldCol, 0, 1);
      }
    }

    // ---- compute on tile t (buffer cur) while DMA is in flight ----
    // S = Q * K^T for two 16-key columns of this 32-key tile
    v8f s0 = {}, s1 = {};
    {
      const int d0 = hsel * 16;
      const v16h kb0 = cat16(*(const v8h*)&Kt[cur][r16][d0],
                             *(const v8h*)&Kt[cur][r16][d0 + 8]);
      const v16h kb1 = cat16(*(const v8h*)&Kt[cur][r16][32 + d0],
                             *(const v8h*)&Kt[cur][r16][32 + d0 + 8]);
      s0 = wmma_f16(qa0, kb0, s0);
      s0 = wmma_f16(qa1, kb1, s0);
      const v16h kc0 = cat16(*(const v8h*)&Kt[cur][16 + r16][d0],
                             *(const v8h*)&Kt[cur][16 + r16][d0 + 8]);
      const v16h kc1 = cat16(*(const v8h*)&Kt[cur][16 + r16][32 + d0],
                             *(const v8h*)&Kt[cur][16 + r16][32 + d0 + 8]);
      s1 = wmma_f16(qa0, kc0, s1);
      s1 = wmma_f16(qa1, kc1, s1);
    }
    const float madd0 = (maskb[t * 32 + r16]      == 0) ? -1e30f : 0.0f;
    const float madd1 = (maskb[t * 32 + 16 + r16] == 0) ? -1e30f : 0.0f;

    // Online softmax update. C layout: per VGPR r, each 16-lane half shares one row,
    // so row reductions need only xor 1/2/4/8 (wave32).
#pragma unroll
    for (int r = 0; r < 8; ++r) {
      float a = s0[r] * 0.125f + madd0;   // 1/sqrt(64)
      float c = s1[r] * 0.125f + madd1;
      float mx = fmaxf(a, c);
#pragma unroll
      for (int off = 1; off < 16; off <<= 1) mx = fmaxf(mx, __shfl_xor(mx, off, 32));
      const float mn = fmaxf(mrow[r], mx);
      const float al = __expf(mrow[r] - mn);
      mrow[r] = mn;
      const float p0 = __expf(a - mn);
      const float p1 = __expf(c - mn);
      float rs = p0 + p1;
#pragma unroll
      for (int off = 1; off < 16; off <<= 1) rs += __shfl_xor(rs, off, 32);
      lrow[r] = lrow[r] * al + rs;
#pragma unroll
      for (int j = 0; j < 4; ++j) O[j][r] *= al;
      const int prow = r + 8 * hsel;
      Pb[wave][prow][r16]      = (_Float16)p0;
      Pb[wave][prow][16 + r16] = (_Float16)p1;
    }

    // wave-private C->A relayout through LDS; enforce store->load ordering
    asm volatile("s_wait_dscnt 0" ::: "memory");

    const v16h pa = cat16(*(const v8h*)&Pb[wave][r16][k0],
                          *(const v8h*)&Pb[wave][r16][8 + k0]);
    const int kv0 = hsel * 16;
#pragma unroll
    for (int j = 0; j < 4; ++j) {
      const v16h vb = cat16(*(const v8h*)&Vt[cur][j * 16 + r16][kv0],
                            *(const v8h*)&Vt[cur][j * 16 + r16][kv0 + 8]);
      O[j] = wmma_f16(pa, vb, O[j]);
    }

    // ---- finish staging tile t+1, then one barrier ----
    if (have_next) {
#pragma unroll
      for (int i = 0; i < 8; ++i) Vt[nxt][ldCol + i][ldRow] = vv[i];
      wait_async_zero();
    }
    __syncthreads();
  }

  // epilogue: normalize and write f32 output [B,T1,1024]
#pragma unroll
  for (int r = 0; r < 8; ++r) {
    const float inv = 1.0f / lrow[r];
    const int t1 = qBase + wave * 16 + r + 8 * hsel;
#pragma unroll
    for (int j = 0; j < 4; ++j) {
      const int f = h * DKH + j * 16 + r16;
      out[((size_t)b * T1L + t1) * NFEAT + f] = O[j][r] * inv;
    }
  }
}

// ---------------------------------------------------------------------------
extern "C" void kernel_launch(void* const* d_in, const int* in_sizes, int n_in,
                              void* d_out, int out_size, void* d_ws, size_t ws_size,
                              hipStream_t stream) {
  const float* query = (const float*)d_in[0];
  const float* key   = (const float*)d_in[1];
  const float* value = (const float*)d_in[2];
  const int*   mask  = (const int*)d_in[3];
  const float* Wq    = (const float*)d_in[4];
  const float* bq    = (const float*)d_in[5];
  float* out = (float*)d_out;

  const size_t per = (size_t)B_SZ * HEADS * T1L * DKH;  // 8,388,608 halves each
  _Float16* qh = (_Float16*)d_ws;                       // projected Q, head-major
  _Float16* kh = qh + per;                              // K f16 head-major
  _Float16* vh = kh + per;                              // V f16 head-major
  _Float16* qx = vh + per;                              // query f16, [8192,1024]
  _Float16* Wh = qx + (size_t)B_SZ * T1L * NFEAT;       // Wq f16, [1024,1024]
  // total workspace: 48 MB + 16 MB + 2 MB = 66 MB

  const int totalQuads = (B_SZ * T2L * NFEAT) / 4;       // 2,097,152
  cvt_head_kernel<<<totalQuads / 256, 256, 0, stream>>>(key,   kh);
  cvt_head_kernel<<<totalQuads / 256, 256, 0, stream>>>(value, vh);
  cvt_plain_kernel<<<totalQuads / 256, 256, 0, stream>>>(query, qx);
  cvt_plain_kernel<<<(NFEAT * NFEAT / 4) / 256, 256, 0, stream>>>(Wq, Wh);

  qproj_kernel<<<dim3((B_SZ * T1L) / 64, NFEAT / 64), 128, 0, stream>>>(qx, Wh, bq, qh);

  attn_kernel<<<dim3(T1L / 128, B_SZ * HEADS), 256, 0, stream>>>(qh, kh, vh, mask, out);
}
